// GraphQNNHybrid_65481071410833
// MI455X (gfx1250) — matrix operand order repaired and checked by reading, not compiled, not run on hardware
//
#include <hip/hip_runtime.h>
#include <hip/hip_bf16.h>
#include <math.h>

typedef __attribute__((ext_vector_type(16))) _Float16 v16h;
typedef __attribute__((ext_vector_type(8)))  float    v8f;

#define NNODES 10000
#define F_IN   128
#define F_HID  256
#define F_OUT  128

// Branch-free activations (keep EXEC full around WMMA, no libm saveexec dance)
__device__ __forceinline__ float tanh_fast(float x) {
    return 1.0f - 2.0f * __builtin_amdgcn_rcpf(__expf(2.0f * x) + 1.0f);
}
__device__ __forceinline__ float sigmoid_fast(float x) {
    return __builtin_amdgcn_rcpf(1.0f + __expf(-x));
}

// ---- Weight pre-pack: f32 row-major W[K][N] -> fragment-major f16 ----------
// packed[((nt*nkf + kf)*32 + lane)*16 + i] = W[kf*32 + (lane>>4)*16 + i][nt*16 + (lane&15)]
// so a B fragment is one contiguous 32B-per-lane load (2x global_load_b128).
__global__ void pack_weights_kernel(const float* __restrict__ W, int K, int N,
                                    _Float16* __restrict__ out)
{
    const int idx = blockIdx.x * blockDim.x + threadIdx.x;
    if (idx >= K * N) return;
    const int i    = idx & 15;
    const int lane = (idx >> 4) & 31;
    const int frag = idx >> 9;
    const int nkf  = K / 32;
    const int kf   = frag % nkf;
    const int nt   = frag / nkf;
    const int k = kf * 32 + (lane >> 4) * 16 + i;
    const int n = nt * 16 + (lane & 15);
    out[idx] = (_Float16)W[(size_t)k * N + n];
}

// ---- Fragment loaders (wave32, v_wmma_f32_16x16x32_f16 layouts) ------------

// A-matrix 16x32 f16 fragment from a row-major LDS tile.
__device__ __forceinline__ v16h load_a_frag_lds(const _Float16* base, int ld,
                                                int k0, int lane) {
    const int row = lane & 15;
    const int sel = lane >> 4;
    const _Float16* p = base + row * ld + k0 + sel * 16;
    v16h a;
#pragma unroll
    for (int i = 0; i < 8; ++i) a[i] = p[i];
#pragma unroll
    for (int i = 0; i < 8; ++i) a[8 + i] = p[8 + i];
    return a;
}

// B fragment: single 32B/lane load from the packed array.
__device__ __forceinline__ v16h load_b_frag(const v16h* __restrict__ P, int nkf,
                                            int nt, int kf, int lane) {
    return P[(size_t)(nt * nkf + kf) * 32 + lane];
}

// ---- Kernel 1: fused MLP encoder  h = sigmoid(tanh(tanh(X W1 + b1) W2 + b2))
__global__ __launch_bounds__(32) void encoder_kernel(
    const float* __restrict__ X,
    const v16h* __restrict__ pW1, const float* __restrict__ b1,
    const v16h* __restrict__ pW2, const float* __restrict__ b2,
    float* __restrict__ hidden)
{
    __shared__ _Float16 sA[16 * F_IN];    // 4 KB: input tile, f16
    __shared__ _Float16 sH[16 * F_HID];   // 8 KB: layer-1 activations, f16
    const int lane = threadIdx.x;
    const int row0 = blockIdx.x * 16;

    for (int i = lane; i < 16 * F_IN; i += 32) {
        sA[i] = (_Float16)X[(size_t)(row0 + (i >> 7)) * F_IN + (i & (F_IN - 1))];
    }
    __syncthreads();

    // ---- layer 1: 16x128 @ 128x256, tanh epilogue into LDS -----------------
    for (int nt = 0; nt < F_HID / 16; ++nt) {
        const int col = nt * 16 + (lane & 15);
        v8f acc = {};
#pragma unroll
        for (int kf = 0; kf < F_IN / 32; ++kf) {
            v16h a = load_a_frag_lds(sA, F_IN, kf * 32, lane);
            v16h b = load_b_frag(pW1, F_IN / 32, nt, kf, lane);
            acc = __builtin_amdgcn_wmma_f32_16x16x32_f16(false, a, false, b,
                                                         (short)0, acc, false, false);
        }
        const float bias = b1[col];
#pragma unroll
        for (int r = 0; r < 8; ++r) {
            const int m = r + 8 * (lane >> 4);
            sH[m * F_HID + col] = (_Float16)tanh_fast(acc[r] + bias);
        }
    }
    __syncthreads();

    // ---- layer 2: 16x256 @ 256x256, tanh+sigmoid epilogue to global --------
    for (int nt = 0; nt < F_HID / 16; ++nt) {
        const int col = nt * 16 + (lane & 15);
        v8f acc = {};
#pragma unroll
        for (int kf = 0; kf < F_HID / 32; ++kf) {
            v16h a = load_a_frag_lds(sH, F_HID, kf * 32, lane);
            v16h b = load_b_frag(pW2, F_HID / 32, nt, kf, lane);
            acc = __builtin_amdgcn_wmma_f32_16x16x32_f16(false, a, false, b,
                                                         (short)0, acc, false, false);
        }
        const float bias = b2[col];
#pragma unroll
        for (int r = 0; r < 8; ++r) {
            const int m = r + 8 * (lane >> 4);
            hidden[(size_t)(row0 + m) * F_HID + col] =
                sigmoid_fast(tanh_fast(acc[r] + bias));
        }
    }
}

// ---- Kernel 2: zero sums + deg (graph-capture-safe init) -------------------
__global__ void zero_ws_kernel(float* __restrict__ sums, float* __restrict__ deg)
{
    const size_t i = (size_t)blockIdx.x * blockDim.x + threadIdx.x;
    if (i < (size_t)NNODES * F_HID) sums[i] = 0.0f;
    if (i < NNODES)                 deg[i]  = 0.0f;
}

// ---- Kernel 3: edge scatter (one wave per edge, L2-resident atomics) -------
__global__ void scatter_kernel(const float* __restrict__ hidden,
                               const int* __restrict__ esrc,
                               const int* __restrict__ edst,
                               float* __restrict__ sums, float* __restrict__ deg,
                               int n_edges)
{
    const int e = blockIdx.x * (blockDim.x >> 5) + (threadIdx.x >> 5);
    if (e >= n_edges) return;
    const int lane = threadIdx.x & 31;
    const int s = esrc[e];
    const int d = edst[e];
    // lane covers 8 contiguous features: two b128 gathers, 8 f32 atomics
    const float4* hs = (const float4*)(hidden + (size_t)s * F_HID) + lane * 2;
    float*        sd = sums + (size_t)d * F_HID + lane * 8;
    const float4 v0 = hs[0];
    const float4 v1 = hs[1];
    atomicAdd(&sd[0], v0.x); atomicAdd(&sd[1], v0.y);
    atomicAdd(&sd[2], v0.z); atomicAdd(&sd[3], v0.w);
    atomicAdd(&sd[4], v1.x); atomicAdd(&sd[5], v1.y);
    atomicAdd(&sd[6], v1.z); atomicAdd(&sd[7], v1.w);
    if (lane == 0) atomicAdd(&deg[d], 1.0f);
}

// ---- Kernel 4: out = (sums / max(deg,1)) @ Wo + bo -------------------------
__global__ __launch_bounds__(32) void output_kernel(
    const float* __restrict__ sums, const float* __restrict__ deg,
    const v16h* __restrict__ pWo, const float* __restrict__ bo,
    float* __restrict__ out)
{
    __shared__ _Float16 sAgg[16 * F_HID];   // 8 KB
    const int lane = threadIdx.x;
    const int row0 = blockIdx.x * 16;

    for (int i = lane; i < 16 * F_HID; i += 32) {
        const int r = i >> 8, c = i & (F_HID - 1);
        const float dg = fmaxf(deg[row0 + r], 1.0f);
        sAgg[i] = (_Float16)(sums[(size_t)(row0 + r) * F_HID + c] *
                             __builtin_amdgcn_rcpf(dg));
    }
    __syncthreads();

    for (int nt = 0; nt < F_OUT / 16; ++nt) {
        const int col = nt * 16 + (lane & 15);
        v8f acc = {};
#pragma unroll
        for (int kf = 0; kf < F_HID / 32; ++kf) {
            v16h a = load_a_frag_lds(sAgg, F_HID, kf * 32, lane);
            v16h b = load_b_frag(pWo, F_HID / 32, nt, kf, lane);
            acc = __builtin_amdgcn_wmma_f32_16x16x32_f16(false, a, false, b,
                                                         (short)0, acc, false, false);
        }
        const float bias = bo[col];
#pragma unroll
        for (int r = 0; r < 8; ++r) {
            const int m = r + 8 * (lane >> 4);
            out[(size_t)(row0 + m) * F_OUT + col] = acc[r] + bias;
        }
    }
}

// ---- Host launcher ---------------------------------------------------------
extern "C" void kernel_launch(void* const* d_in, const int* in_sizes, int n_in,
                              void* d_out, int out_size, void* d_ws, size_t ws_size,
                              hipStream_t stream)
{
    const float* X    = (const float*)d_in[0];
    const int*   esrc = (const int*)  d_in[1];
    const int*   edst = (const int*)  d_in[2];
    const float* W1   = (const float*)d_in[3];
    const float* b1   = (const float*)d_in[4];
    const float* W2   = (const float*)d_in[5];
    const float* b2   = (const float*)d_in[6];
    const float* Wo   = (const float*)d_in[7];
    const float* bo   = (const float*)d_in[8];
    float* out = (float*)d_out;
    const int n_edges = in_sizes[1];

    char* ws = (char*)d_ws;
    float* hidden = (float*)ws;                                   // N*256 f32
    float* sums   = (float*)(ws + (size_t)NNODES * F_HID * 4);    // N*256 f32
    float* deg    = (float*)(ws + 2ull * NNODES * F_HID * 4);     // N f32
    char*  pk     = ws + 2ull * NNODES * F_HID * 4 + NNODES * 4;  // 32B aligned
    _Float16* pW1 = (_Float16*)pk;                                // 64 KB
    _Float16* pW2 = pW1 + F_IN * F_HID;                           // 128 KB
    _Float16* pWo = pW2 + F_HID * F_HID;                          // 64 KB

    // 1) pack weights into WMMA fragment-major f16 (tiny, one-shot per launch)
    pack_weights_kernel<<<(F_IN * F_HID + 255) / 256, 256, 0, stream>>>(W1, F_IN,  F_HID, pW1);
    pack_weights_kernel<<<(F_HID * F_HID + 255) / 256, 256, 0, stream>>>(W2, F_HID, F_HID, pW2);
    pack_weights_kernel<<<(F_HID * F_OUT + 255) / 256, 256, 0, stream>>>(Wo, F_HID, F_OUT, pWo);

    const int tiles = NNODES / 16;  // 625

    encoder_kernel<<<tiles, 32, 0, stream>>>(X, (const v16h*)pW1, b1,
                                             (const v16h*)pW2, b2, hidden);

    const size_t total = (size_t)NNODES * F_HID;
    zero_ws_kernel<<<(int)((total + 255) / 256), 256, 0, stream>>>(sums, deg);

    scatter_kernel<<<(n_edges + 7) / 8, 256, 0, stream>>>(hidden, esrc, edst,
                                                          sums, deg, n_edges);

    output_kernel<<<tiles, 32, 0, stream>>>(sums, deg, (const v16h*)pWo, bo, out);
}
